// QRNN_34626026341009
// MI455X (gfx1250) — matrix operand order repaired and checked by reading, not compiled, not run on hardware
//
#include <hip/hip_runtime.h>

// Problem constants (fixed by the reference):
// B=16, T=4096, D_IN=H=512, L=2, 3H=1536, M=B*T=65536
#define QM      65536      // rows of GEMM (B*T)
#define QK      512        // inner dim
#define QN      1536       // 3*H
#define QH      512
#define QB      16
#define QT      4096
#define QBH     8192       // B*H channels
#define QNCH    16         // scan chunks
#define QCHUNK  256        // T / QNCH
#define QG      33554432ull // B*T*H elements
#define QWN     786432     // 1536*512 weight elements

#if defined(__has_builtin)
#if __has_builtin(__builtin_amdgcn_global_load_async_to_lds_b128)
#define QRNN_ASYNC_LDS 1
#endif
#endif

typedef __attribute__((ext_vector_type(16))) __bf16 v16bf;
typedef __attribute__((ext_vector_type(8)))  float  v8f;
typedef __attribute__((ext_vector_type(4)))  int    v4i;

union BV16 { uint4 q[2]; v16bf v; };   // 32 bytes: 16 bf16 halves

__device__ __forceinline__ unsigned short bf16_rne(float x) {
  unsigned int u = __float_as_uint(x);
  unsigned int r = u + 0x7FFFu + ((u >> 16) & 1u);
  return (unsigned short)(r >> 16);
}
__device__ __forceinline__ float bf16_to_f(unsigned short h) {
  return __uint_as_float(((unsigned int)h) << 16);
}
__device__ __forceinline__ float sigmoidf(float x) {
  return 1.0f / (1.0f + __expf(-x));
}
// split one fp32 into bf16 hi (RNE) + bf16 lo (RNE of residual)
__device__ __forceinline__ void bf16_split(float x, unsigned short& hi,
                                           unsigned short& lo) {
  hi = bf16_rne(x);
  lo = bf16_rne(x - bf16_to_f(hi));
}

// ---------------------------------------------------------------------------
// Prepass: split W[N,K] fp32 into persistent bf16 hi/lo planes (L2-resident).
// ---------------------------------------------------------------------------
__global__ __launch_bounds__(256) void qrnn_wsplit(
    const float* __restrict__ W, unsigned short* __restrict__ Wh,
    unsigned short* __restrict__ Wl) {
  const int gid = blockIdx.x * 256 + threadIdx.x;   // 0..QWN-1
  unsigned short h, l;
  bf16_split(W[gid], h, l);
  Wh[gid] = h;
  Wl[gid] = l;
}

// ---------------------------------------------------------------------------
// Kernel 1: gates = act( X[M,K] * W[N,K]^T + b ) -> z'(relu), f'(sig), o'(sig)
// bf16x3 split-precision WMMA: D += Ah*Bh + Ah*Bl + Al*Bh (f32 accumulate).
// Block: 256 threads = 8 waves; tile M=32 x N=64; each wave one 16x16 D tile.
// LDS holds bf16 hi/lo planes; per-lane operand gather = pure ds_load_b128.
// B planes stream global->LDS via GLOBAL_LOAD_ASYNC_TO_LDS_B128 (ASYNCcnt).
// ---------------------------------------------------------------------------
__global__ __launch_bounds__(256) void qrnn_gates_wmma(
    const float* __restrict__ X, const unsigned short* __restrict__ Wh,
    const unsigned short* __restrict__ Wl, const float* __restrict__ bias,
    float* __restrict__ zp, float* __restrict__ fp, float* __restrict__ op) {
  // rows padded to 40 halves (80B): keeps 16B alignment, kills bank conflicts
  __shared__ unsigned short Ah[32][40];
  __shared__ unsigned short Al[32][40];
  __shared__ unsigned short Bh[64][40];
  __shared__ unsigned short Bl[64][40];

  const int tid  = threadIdx.x;
  const int lane = tid & 31;
  const int w    = tid >> 5;
  const int mi   = w >> 2;             // 0..1  (M sub-tile)
  const int ni   = w & 3;              // 0..3  (N sub-tile)
  const int m0   = blockIdx.y * 32;
  const int n0   = blockIdx.x * 64;

  // staging coordinates
  const int ar = tid >> 3;             // A: row 0..31
  const int ac = (tid & 7) * 4;        // A: col (floats) 0..28
  const int br = tid >> 2;             // B: row 0..63
  const int bc = (tid & 3) * 8;        // B: col (halves) 0..24

  // per-lane gather coordinates (WMMA operand layouts)
  const int arow = mi * 16 + (lane & 15);
  const int brow = ni * 16 + (lane & 15);
  const int ka   = (lane >> 4) * 8;    // A halves: [ka..ka+7], [ka+16..ka+23]
  const int kb   = (lane >> 4) * 8;    // B halves: [kb..kb+7], [kb+16..kb+23]

  v8f acc = {};

  for (int k0 = 0; k0 < QK; k0 += 32) {
    __syncthreads();  // WAR guard vs previous iteration's operand reads

    // ---- stage B: bf16 copy from pre-split planes ----
    const unsigned short* gbh = &Wh[(size_t)(n0 + br) * QK + k0 + bc];
    const unsigned short* gbl = &Wl[(size_t)(n0 + br) * QK + k0 + bc];
#if defined(QRNN_ASYNC_LDS)
    typedef __attribute__((address_space(1))) v4i* gv4p;  // global int4*
    typedef __attribute__((address_space(3))) v4i* lv4p;  // LDS int4*
    __builtin_amdgcn_global_load_async_to_lds_b128((gv4p)gbh,
                                                   (lv4p)&Bh[br][bc], 0, 0);
    __builtin_amdgcn_global_load_async_to_lds_b128((gv4p)gbl,
                                                   (lv4p)&Bl[br][bc], 0, 0);
#else
    *(uint4*)&Bh[br][bc] = *(const uint4*)gbh;
    *(uint4*)&Bl[br][bc] = *(const uint4*)gbl;
#endif

    // ---- stage A: fp32 -> bf16 hi/lo, converted ONCE per block ----
    const float4 xa = *(const float4*)&X[(size_t)(m0 + ar) * QK + k0 + ac];
    unsigned short h0, l0, h1, l1, h2, l2, h3, l3;
    bf16_split(xa.x, h0, l0);
    bf16_split(xa.y, h1, l1);
    bf16_split(xa.z, h2, l2);
    bf16_split(xa.w, h3, l3);
    uint2 hp, lp;
    hp.x = ((unsigned)h1 << 16) | h0;  hp.y = ((unsigned)h3 << 16) | h2;
    lp.x = ((unsigned)l1 << 16) | l0;  lp.y = ((unsigned)l3 << 16) | l2;
    *(uint2*)&Ah[ar][ac] = hp;
    *(uint2*)&Al[ar][ac] = lp;

    if (k0 + 32 < QK) {  // pull next X slab into near caches (WGP scope)
      __builtin_prefetch(&X[(size_t)(m0 + ar) * QK + k0 + 32 + ac], 0, 3);
    }

#if defined(QRNN_ASYNC_LDS)
    // __syncthreads covers DScnt only; async LDS writes ride ASYNCcnt.
#if __has_builtin(__builtin_amdgcn_s_wait_asynccnt)
    __builtin_amdgcn_s_wait_asynccnt(0);
#else
    asm volatile("s_wait_asynccnt 0" ::: "memory");
#endif
#endif
    __syncthreads();

    // ---- per-lane operand gather: contiguous 16B runs -> ds_load_b128 ----
    BV16 va_h, va_l, vb_h, vb_l;
    va_h.q[0] = *(const uint4*)&Ah[arow][ka];
    va_h.q[1] = *(const uint4*)&Ah[arow][ka + 16];
    va_l.q[0] = *(const uint4*)&Al[arow][ka];
    va_l.q[1] = *(const uint4*)&Al[arow][ka + 16];
    vb_h.q[0] = *(const uint4*)&Bh[brow][kb];
    vb_h.q[1] = *(const uint4*)&Bh[brow][kb + 16];
    vb_l.q[0] = *(const uint4*)&Bl[brow][kb];
    vb_l.q[1] = *(const uint4*)&Bl[brow][kb + 16];

    // bf16x3 product, fp32 accumulate
    acc = __builtin_amdgcn_wmma_f32_16x16x32_bf16(false, va_h.v, false, vb_h.v,
                                                  (short)0, acc, false, false);
    acc = __builtin_amdgcn_wmma_f32_16x16x32_bf16(false, va_h.v, false, vb_l.v,
                                                  (short)0, acc, false, false);
    acc = __builtin_amdgcn_wmma_f32_16x16x32_bf16(false, va_l.v, false, vb_h.v,
                                                  (short)0, acc, false, false);
  }

  // Epilogue: D layout = VGPR v, lane l -> row v + 8*(l>>4), col (l&15).
  const int n_glob = n0 + ni * 16 + (lane & 15);
  const float bv   = bias[n_glob];
  const int gate   = n_glob >> 9;      // 0=Z(relu) 1=F(sigmoid) 2=O(sigmoid)
  const int col    = n_glob & 511;
  float* dst = (gate == 0) ? zp : (gate == 1) ? fp : op;
#pragma unroll
  for (int v = 0; v < 8; ++v) {
    const int m = m0 + mi * 16 + v + 8 * (lane >> 4);
    float g = acc[v] + bv;
    g = (gate == 0) ? fmaxf(g, 0.0f) : sigmoidf(g);
    dst[(size_t)m * QH + col] = g;
  }
}

// ---------------------------------------------------------------------------
// Chunked linear-recurrence scan: c_t = f_t*c_{t-1} + (1-f_t)*z_t
// Phase 1: per (channel, chunk) compose affine map (A, B): c_out = A*c_in + B
// ---------------------------------------------------------------------------
__global__ __launch_bounds__(256) void qrnn_chunk_reduce(
    const float* __restrict__ fp, const float* __restrict__ zp,
    float* __restrict__ Pw, float* __restrict__ Qw) {
  const int gid   = blockIdx.x * 256 + threadIdx.x;  // 0..QBH*QNCH-1
  const int chan  = gid & (QBH - 1);
  const int chunk = gid >> 13;          // gid / QBH
  const int b = chan >> 9, h = chan & 511;
  size_t idx = ((size_t)b * QT + (size_t)chunk * QCHUNK) * QH + h;
  float A = 1.0f, Bv = 0.0f;
#pragma unroll 4
  for (int t = 0; t < QCHUNK; ++t) {
    const float f = fp[idx];
    const float z = zp[idx];
    A *= f;
    Bv = fmaf(f, Bv - z, z);            // f*Bv + (1-f)*z
    idx += QH;
  }
  Pw[(size_t)chunk * QBH + chan] = A;
  Qw[(size_t)chunk * QBH + chan] = Bv;
}

// Phase 2: tiny sequential scan across the 16 chunks; emits chunk-entry
// states and the final hidden state h[:, layer, :].
__global__ __launch_bounds__(256) void qrnn_chunk_scan(
    const float* __restrict__ h0, const float* __restrict__ Pw,
    const float* __restrict__ Qw, float* __restrict__ Cs,
    float* __restrict__ hout, int layer) {
  const int chan = blockIdx.x * 256 + threadIdx.x;   // 0..QBH-1
  const int b = chan >> 9, h = chan & 511;
  float c = h0[(size_t)b * 1024 + (size_t)layer * QH + h];
#pragma unroll
  for (int ch = 0; ch < QNCH; ++ch) {
    Cs[(size_t)ch * QBH + chan] = c;                 // state entering chunk
    c = fmaf(Pw[(size_t)ch * QBH + chan], c, Qw[(size_t)ch * QBH + chan]);
  }
  hout[(size_t)b * 1024 + (size_t)layer * QH + h] = c;
}

// Phase 3: replay each chunk from its entry state; x_out = o' * c_t
__global__ __launch_bounds__(256) void qrnn_chunk_apply(
    const float* __restrict__ fp, const float* __restrict__ zp,
    const float* __restrict__ op, const float* __restrict__ Cs,
    float* __restrict__ xout) {
  const int gid   = blockIdx.x * 256 + threadIdx.x;
  const int chan  = gid & (QBH - 1);
  const int chunk = gid >> 13;
  const int b = chan >> 9, h = chan & 511;
  float c = Cs[(size_t)chunk * QBH + chan];
  size_t idx = ((size_t)b * QT + (size_t)chunk * QCHUNK) * QH + h;
#pragma unroll 4
  for (int t = 0; t < QCHUNK; ++t) {
    const float f = fp[idx];
    const float z = zp[idx];
    c = fmaf(f, c - z, z);              // f*c + (1-f)*z
    xout[idx] = op[idx] * c;
    idx += QH;
  }
}

// ---------------------------------------------------------------------------
extern "C" void kernel_launch(void* const* d_in, const int* in_sizes, int n_in,
                              void* d_out, int out_size, void* d_ws,
                              size_t ws_size, hipStream_t stream) {
  (void)in_sizes; (void)n_in; (void)out_size; (void)ws_size;
  const float* x  = (const float*)d_in[0];   // [16,4096,512]
  const float* h0 = (const float*)d_in[1];   // [16,2,512]
  const float* W0 = (const float*)d_in[2];   // [1536,512]
  const float* b0 = (const float*)d_in[3];   // [1536]
  const float* W1 = (const float*)d_in[4];   // [1536,512]
  const float* b1 = (const float*)d_in[5];   // [1536]

  float* xout = (float*)d_out;               // [B*T*512]
  float* hout = xout + QG;                   // [16,2,512]

  // Workspace: z' f' o' gate planes (fp32) + scan scratch + bf16 W planes.
  float* zp = (float*)d_ws;
  float* fp = zp + QG;
  float* op = fp + QG;
  float* Pw = op + QG;
  float* Qw = Pw + (size_t)QBH * QNCH;
  float* Cs = Qw + (size_t)QBH * QNCH;
  unsigned short* Wh0 = (unsigned short*)(Cs + (size_t)QBH * QNCH);
  unsigned short* Wl0 = Wh0 + QWN;
  unsigned short* Wh1 = Wl0 + QWN;
  unsigned short* Wl1 = Wh1 + QWN;

  const dim3 gemm_grid(QN / 64, QM / 32);    // (24, 2048)
  const int  sgrid = (QBH * QNCH) / 256;     // 512 blocks

  // one-shot weight splits (3 MB each, stays L2-resident)
  qrnn_wsplit<<<QWN / 256, 256, 0, stream>>>(W0, Wh0, Wl0);
  qrnn_wsplit<<<QWN / 256, 256, 0, stream>>>(W1, Wh1, Wl1);

  // ---- layer 0 ----
  qrnn_gates_wmma<<<gemm_grid, 256, 0, stream>>>(x, Wh0, Wl0, b0, zp, fp, op);
  qrnn_chunk_reduce<<<sgrid, 256, 0, stream>>>(fp, zp, Pw, Qw);
  qrnn_chunk_scan<<<QBH / 256, 256, 0, stream>>>(h0, Pw, Qw, Cs, hout, 0);
  qrnn_chunk_apply<<<sgrid, 256, 0, stream>>>(fp, zp, op, Cs, xout);

  // ---- layer 1 (reads layer-0 x from d_out, overwrites it with final x) ----
  qrnn_gates_wmma<<<gemm_grid, 256, 0, stream>>>(xout, Wh1, Wl1, b1, zp, fp, op);
  qrnn_chunk_reduce<<<sgrid, 256, 0, stream>>>(fp, zp, Pw, Qw);
  qrnn_chunk_scan<<<QBH / 256, 256, 0, stream>>>(h0, Pw, Qw, Cs, hout, 1);
  qrnn_chunk_apply<<<sgrid, 256, 0, stream>>>(fp, zp, op, Cs, xout);
}
